// MultiHeadAttentionQuantum_65481071400587
// MI455X (gfx1250) — matrix operand order repaired and checked by reading, not compiled, not run on hardware
//
#include <hip/hip_runtime.h>

typedef __attribute__((ext_vector_type(16))) _Float16 v16h;
typedef __attribute__((ext_vector_type(4)))  _Float16 v4h;
typedef __attribute__((ext_vector_type(8)))  float    v8f;

#define SEQ     2048
#define NCHUNK  8          // query chunks per batch
#define QLOC    256        // queries handled per workgroup
#define PSTRIDE 36         // padded row stride (dwords) of P tile -> bank-conflict free
#define VPAD    2064       // padded hkT row length (halves), keeps 16B alignment

__global__ __launch_bounds__(256)
void qattn_fused(const float* __restrict__ x,
                 const float* __restrict__ theta,
                 const float* __restrict__ Wout,
                 float* __restrict__ out)
{
  // h as f16, key-major [head][k][d] and transposed [head][d][k]
  __shared__ alignas(16) _Float16 hk [2][SEQ][4];        // 32 KB
  __shared__ alignas(16) _Float16 hkT[2][4][VPAD];       // ~33 KB
  __shared__ alignas(16) float    Pbuf[8][16 * PSTRIDE]; // 18 KB (per-wave P tiles)
  __shared__ alignas(16) float    attn[QLOC][8];         // 8 KB pre-projection output
  __shared__ float wlds[64];

  const int  tid   = threadIdx.x;
  const int  wave  = tid >> 5;
  const int  lane  = tid & 31;
  const int  lh    = lane & 15;
  const bool hi    = lane >= 16;
  const int  b     = blockIdx.x / NCHUNK;
  const int  chunk = blockIdx.x % NCHUNK;

  if (tid < 64) wlds[tid] = Wout[tid];

  float4 t0 = *(const float4*)theta;
  float4 t1 = *(const float4*)(theta + 4);

  // ---- Phase 1: h = cos(x + theta) for the whole batch row, into LDS (f16) ----
  const float4* xp = (const float4*)(x + (size_t)b * SEQ * 8);
#pragma unroll
  for (int i = 0; i < 16; ++i) {
    int idx  = tid + i * 256;        // float4 index within batch: = s*2 + head
    int s    = idx >> 1;
    int head = idx & 1;
    float4 xv = xp[idx];
    float4 tt = head ? t1 : t0;
    v4h hv;
    hv[0] = (_Float16)__cosf(xv.x + tt.x);
    hv[1] = (_Float16)__cosf(xv.y + tt.y);
    hv[2] = (_Float16)__cosf(xv.z + tt.z);
    hv[3] = (_Float16)__cosf(xv.w + tt.w);
    *(v4h*)&hk[head][s][0] = hv;
    hkT[head][0][s] = hv[0];
    hkT[head][1][s] = hv[1];
    hkT[head][2][s] = hv[2];
    hkT[head][3][s] = hv[3];
  }
  __syncthreads();

  // ---- Phase 2: fused attention; each wave: 4 tasks of (head, 16-query block) ----
#pragma unroll 1
  for (int ti = 0; ti < 4; ++ti) {
    const int t    = wave * 4 + ti;       // 0..31
    const int head = t & 1;
    const int qbl  = t >> 1;              // 0..15 local query block
    const int q0   = chunk * QLOC + qbl * 16;

    // A-frag: 16 queries, K=0..3 = head dims (pre-scaled by 1/sqrt(d_k)=0.5), rest 0
    v16h aF = {};
    if (!hi) {
      v4h qv = *(const v4h*)&hk[head][q0 + lh][0];
#pragma unroll
      for (int d = 0; d < 4; ++d) aF[d] = qv[d] * (_Float16)0.5f;
    }

    v8f   acc  = {};                       // cols 0..3 = sum(P*V), col 4 = sum(P) (denom)
    float* prow = &Pbuf[wave][0];

#pragma unroll 1
    for (int kb = 0; kb < 64; ++kb) {
      const int k0 = kb * 32;

      // B-frags for scores: keys k0..k0+15 and k0+16..k0+31, K=0..3 = dims
      v16h b0 = {}, b1 = {};
      if (!hi) {
        v4h kv0 = *(const v4h*)&hk[head][k0 + lh][0];
        v4h kv1 = *(const v4h*)&hk[head][k0 + 16 + lh][0];
#pragma unroll
        for (int d = 0; d < 4; ++d) { b0[d] = kv0[d]; b1[d] = kv1[d]; }
      }
      v8f z  = {};
      v8f s0 = __builtin_amdgcn_wmma_f32_16x16x32_f16(false, aF, false, b0, (short)0, z, false, false);
      v8f s1 = __builtin_amdgcn_wmma_f32_16x16x32_f16(false, aF, false, b1, (short)0, z, false, false);

      // exp (scores bounded in [-2,2] -> no max subtraction needed); stage P in LDS
      const int M0 = hi ? 8 : 0;
#pragma unroll
      for (int r = 0; r < 8; ++r) {
        prow[(r + M0) * PSTRIDE + lh]      = __expf(s0[r]);
        prow[(r + M0) * PSTRIDE + 16 + lh] = __expf(s1[r]);
      }
      __builtin_amdgcn_wave_barrier();

      // read P back in A-layout (row = lane&15; K halves 0..7 -> cols cb..cb+7, 8..15 -> cols 16+cb..)
      const int rb = lh * PSTRIDE + (hi ? 8 : 0);
      float4 pa = *(const float4*)&prow[rb];
      float4 pb = *(const float4*)&prow[rb + 4];
      float4 pc = *(const float4*)&prow[rb + 16];
      float4 pd = *(const float4*)&prow[rb + 20];
      v16h pF;
      pF[0]=(_Float16)pa.x;  pF[1]=(_Float16)pa.y;  pF[2]=(_Float16)pa.z;  pF[3]=(_Float16)pa.w;
      pF[4]=(_Float16)pb.x;  pF[5]=(_Float16)pb.y;  pF[6]=(_Float16)pb.z;  pF[7]=(_Float16)pb.w;
      pF[8]=(_Float16)pc.x;  pF[9]=(_Float16)pc.y;  pF[10]=(_Float16)pc.z; pF[11]=(_Float16)pc.w;
      pF[12]=(_Float16)pd.x; pF[13]=(_Float16)pd.y; pF[14]=(_Float16)pd.z; pF[15]=(_Float16)pd.w;

      // V-frag: N=0..3 -> head dims from transposed h, N=4 -> ones (denominator column)
      v16h vF = {};
      if (lh < 4) {
        const _Float16* vp = &hkT[head][lh][k0 + (hi ? 8 : 0)];
#pragma unroll
        for (int j = 0; j < 16; ++j) vF[j] = vp[j < 8 ? j : j + 8];
      } else if (lh == 4) {
#pragma unroll
        for (int j = 0; j < 16; ++j) vF[j] = (_Float16)1.0f;
      }

      acc = __builtin_amdgcn_wmma_f32_16x16x32_f16(false, pF, false, vF, (short)0, acc, false, false);
      __builtin_amdgcn_wave_barrier();
    }

    // normalize: denom sits in output column N=4 of each half-wave
    const int srcl = (hi ? 16 : 0) + 4;
#pragma unroll
    for (int r = 0; r < 8; ++r) {
      float den = __shfl(acc[r], srcl, 32);
      if (lh < 4) attn[qbl * 16 + r + (hi ? 8 : 0)][head * 4 + lh] = acc[r] / den;
    }
  }
  __syncthreads();

  // ---- Phase 3: output projection y = attn @ W^T, fp32 stores ----
  float a[8];
#pragma unroll
  for (int e = 0; e < 8; ++e) a[e] = attn[tid][e];
  float yy[8];
#pragma unroll
  for (int e = 0; e < 8; ++e) {
    float sacc = 0.f;
#pragma unroll
    for (int k = 0; k < 8; ++k) sacc += a[k] * wlds[e * 8 + k];
    yy[e] = sacc;
  }
  float4* op = (float4*)(out + ((size_t)b * SEQ + chunk * QLOC + tid) * 8);
  op[0] = make_float4(yy[0], yy[1], yy[2], yy[3]);
  op[1] = make_float4(yy[4], yy[5], yy[6], yy[7]);
}

extern "C" void kernel_launch(void* const* d_in, const int* in_sizes, int n_in,
                              void* d_out, int out_size, void* d_ws, size_t ws_size,
                              hipStream_t stream)
{
  const float* x     = (const float*)d_in[0];   // [32, 2048, 8]
  const float* theta = (const float*)d_in[1];   // [8]
  const float* Wout  = (const float*)d_in[2];   // [8, 8]
  float* out = (float*)d_out;                   // [32, 2048, 8]
  dim3 grid(32 * NCHUNK), block(256);
  hipLaunchKernelGGL(qattn_fused, grid, block, 0, stream, x, theta, Wout, out);
}
